// BiasFilter_773094113754
// MI455X (gfx1250) — matrix operand order repaired and checked by reading, not compiled, not run on hardware
//
#include <hip/hip_runtime.h>
#include <hip/hip_bf16.h>
#include <math.h>

// CDNA5 / gfx1250: wave32, WMMA 16x16x32 bf16 -> f32 accum.
typedef __bf16 bf16_t;
typedef __attribute__((ext_vector_type(16))) __bf16 v16bf;
typedef __attribute__((ext_vector_type(8)))  float  v8f;

#define LN_EPS 1e-5f
#define NBIAS  10
#define E_DIM  1024
#define H_DIM  512
#define TM     32      // rows per workgroup (two 16-row m-tiles, B reuse x2)

__device__ __forceinline__ float gelu_erf(float v) {
    // exact-erf GELU (torch nn.GELU default)
    return 0.5f * v * (1.0f + erff(v * 0.70710678118654752f));
}

// ---------------- Kernel P: one-time f32 -> bf16 weight conversion ----------------
__global__ __launch_bounds__(256) void cvt_weights(
    const float* __restrict__ W1, const float* __restrict__ W2,
    bf16_t* __restrict__ W1b, bf16_t* __restrict__ W2b)
{
    int i = blockIdx.x * 256 + threadIdx.x;
    if (i < E_DIM * E_DIM) W1b[i] = (bf16_t)W1[i];
    if (i < H_DIM * E_DIM) W2b[i] = (bf16_t)W2[i];
}

// ---------------- Kernel A: h = gelu(layernorm(x @ W1^T + b1)) -> bf16 ----------------
// WG = 256 threads (8 waves). Tile: 32 rows x 1024 cols. Each wave: 8 n-tiles x 2 m-tiles.
__global__ __launch_bounds__(256) void gemm1_ln_gelu(
    const float* __restrict__ x, const bf16_t* __restrict__ W1b,
    const float* __restrict__ b1, const float* __restrict__ lng,
    const float* __restrict__ lnb, bf16_t* __restrict__ hout)
{
    extern __shared__ char smem[];
    bf16_t* As = (bf16_t*)smem;          // 32 x 1024 bf16 = 64 KB (dynamic LDS)
    __shared__ float redS[TM], redQ[TM];

    const int tid  = threadIdx.x;
    const int wave = tid >> 5;
    const int lane = tid & 31;
    const int r    = lane & 15;
    const int hf   = lane >> 4;
    const size_t m0 = (size_t)blockIdx.x * TM;

    // Stage x rows into LDS as bf16 (coalesced float4 reads)
    const float4* xsrc = (const float4*)(x + m0 * E_DIM);
    #pragma unroll
    for (int it = 0; it < 32; ++it) {
        int idx = tid + it * 256;              // 8192 float4 = 32x1024 floats
        float4 v = xsrc[idx];
        As[idx * 4 + 0] = (bf16_t)v.x;
        As[idx * 4 + 1] = (bf16_t)v.y;
        As[idx * 4 + 2] = (bf16_t)v.z;
        As[idx * 4 + 3] = (bf16_t)v.w;
    }
    if (tid < TM) { redS[tid] = 0.f; redQ[tid] = 0.f; }
    __syncthreads();

    v8f acc[2][8];
    #pragma unroll
    for (int mt = 0; mt < 2; ++mt)
        #pragma unroll
        for (int p = 0; p < 8; ++p)
            #pragma unroll
            for (int v = 0; v < 8; ++v) acc[mt][p][v] = 0.f;

    const int n_base = wave * 128;

    for (int k0 = 0; k0 < E_DIM; k0 += 32) {
        // A operands: lane r+16h holds row r (m-tile 0) / row 16+r (m-tile 1),
        // K slots {8h+j} and {16+8h+j}
        v16bf a0, a1;
        {
            const bf16_t* ar0 = &As[r * E_DIM + k0 + 8 * hf];
            const bf16_t* ar1 = &As[(16 + r) * E_DIM + k0 + 8 * hf];
            #pragma unroll
            for (int j = 0; j < 8; ++j) { a0[j] = ar0[j];      a1[j] = ar1[j]; }
            #pragma unroll
            for (int j = 0; j < 8; ++j) { a0[8 + j] = ar0[16 + j]; a1[8 + j] = ar1[16 + j]; }
        }
        #pragma unroll
        for (int p = 0; p < 8; ++p) {
            // B operand: lane r+16h holds column n, K slots {8h+j} and {16+8h+j}
            const bf16_t* brow = &W1b[(size_t)(n_base + p * 16 + r) * E_DIM + k0 + 8 * hf];
            v16bf b;
            #pragma unroll
            for (int j = 0; j < 8; ++j) { b[j] = brow[j]; b[8 + j] = brow[16 + j]; }
            acc[0][p] = __builtin_amdgcn_wmma_f32_16x16x32_bf16(
                false, a0, false, b, (short)0, acc[0][p], false, false);
            acc[1][p] = __builtin_amdgcn_wmma_f32_16x16x32_bf16(
                false, a1, false, b, (short)0, acc[1][p], false, false);
        }
    }

    // Epilogue: + b1, row mean/var reduction.
    // Lane holds rows m = 16*mt + 8*hf + v at n = n_base + 16p + r.
    float s1[2][8], s2[2][8];
    #pragma unroll
    for (int mt = 0; mt < 2; ++mt)
        #pragma unroll
        for (int v = 0; v < 8; ++v) { s1[mt][v] = 0.f; s2[mt][v] = 0.f; }
    #pragma unroll
    for (int p = 0; p < 8; ++p) {
        float bias = b1[n_base + p * 16 + r];
        #pragma unroll
        for (int mt = 0; mt < 2; ++mt)
            #pragma unroll
            for (int v = 0; v < 8; ++v) {
                float t = acc[mt][p][v] + bias;
                acc[mt][p][v] = t;
                s1[mt][v] += t; s2[mt][v] += t * t;
            }
    }
    // reduce over the 16 lanes of each half-wave (same row set per half)
    #pragma unroll
    for (int off = 1; off < 16; off <<= 1) {
        #pragma unroll
        for (int mt = 0; mt < 2; ++mt)
            #pragma unroll
            for (int v = 0; v < 8; ++v) {
                s1[mt][v] += __shfl_xor(s1[mt][v], off, 32);
                s2[mt][v] += __shfl_xor(s2[mt][v], off, 32);
            }
    }
    if (r == 0) {
        #pragma unroll
        for (int mt = 0; mt < 2; ++mt)
            #pragma unroll
            for (int v = 0; v < 8; ++v) {
                atomicAdd(&redS[16 * mt + 8 * hf + v], s1[mt][v]);   // ds_add_f32
                atomicAdd(&redQ[16 * mt + 8 * hf + v], s2[mt][v]);
            }
    }
    __syncthreads();

    float mu[2][8], rs[2][8];
    #pragma unroll
    for (int mt = 0; mt < 2; ++mt)
        #pragma unroll
        for (int v = 0; v < 8; ++v) {
            int m = 16 * mt + 8 * hf + v;
            float m1 = redS[m] * (1.f / E_DIM);
            float var = redQ[m] * (1.f / E_DIM) - m1 * m1;
            mu[mt][v] = m1;
            rs[mt][v] = rsqrtf(var + LN_EPS);
        }
    #pragma unroll
    for (int p = 0; p < 8; ++p) {
        int n = n_base + p * 16 + r;
        float gn = lng[n], bn = lnb[n];
        #pragma unroll
        for (int mt = 0; mt < 2; ++mt)
            #pragma unroll
            for (int v = 0; v < 8; ++v) {
                float t = (acc[mt][p][v] - mu[mt][v]) * rs[mt][v] * gn + bn;
                hout[(m0 + 16 * mt + 8 * hf + v) * E_DIM + n] = (bf16_t)gelu_erf(t);
            }
    }
}

// ---------------- Kernel B: h2 = gelu(h @ W2^T + b2); scores; quaternion apply ----------------
// WG = 256 threads (8 waves). Tile: 32 rows x 512 cols. Fully fused epilogue -> out.
__global__ __launch_bounds__(256) void gemm2_scores_quat(
    const bf16_t* __restrict__ hbuf, const bf16_t* __restrict__ W2b,
    const float* __restrict__ b2, const float* __restrict__ W3,
    const float* __restrict__ b3, const float* __restrict__ quats,
    const float* __restrict__ thr, const float* __restrict__ x,
    float* __restrict__ out)
{
    extern __shared__ char smem[];
    bf16_t* As = (bf16_t*)smem;            // 64 KB staged h tile; reused as f32 h2[32][512]
    __shared__ float W3s[NBIAS * H_DIM];   // 20 KB
    __shared__ float QS[TM][4];
    __shared__ int   mrow[TM];

    const int tid  = threadIdx.x;
    const int wave = tid >> 5;
    const int lane = tid & 31;
    const int r    = lane & 15;
    const int hf   = lane >> 4;
    const size_t m0 = (size_t)blockIdx.x * TM;

    // Stage h tile (already bf16) into LDS: 64 KB as uint2 copies
    const uint2* hsrc = (const uint2*)(hbuf + m0 * E_DIM);
    uint2* hdst = (uint2*)As;
    #pragma unroll
    for (int it = 0; it < 32; ++it) hdst[tid + it * 256] = hsrc[tid + it * 256];
    for (int i = tid; i < NBIAS * H_DIM; i += 256) W3s[i] = W3[i];
    if (tid < TM) mrow[tid] = 0;
    __syncthreads();

    v8f acc[2][4];
    #pragma unroll
    for (int mt = 0; mt < 2; ++mt)
        #pragma unroll
        for (int p = 0; p < 4; ++p)
            #pragma unroll
            for (int v = 0; v < 8; ++v) acc[mt][p][v] = 0.f;

    const int n_base = wave * 64;

    for (int k0 = 0; k0 < E_DIM; k0 += 32) {
        v16bf a0, a1;
        {
            const bf16_t* ar0 = &As[r * E_DIM + k0 + 8 * hf];
            const bf16_t* ar1 = &As[(16 + r) * E_DIM + k0 + 8 * hf];
            #pragma unroll
            for (int j = 0; j < 8; ++j) { a0[j] = ar0[j];      a1[j] = ar1[j]; }
            #pragma unroll
            for (int j = 0; j < 8; ++j) { a0[8 + j] = ar0[16 + j]; a1[8 + j] = ar1[16 + j]; }
        }
        #pragma unroll
        for (int p = 0; p < 4; ++p) {
            const bf16_t* brow = &W2b[(size_t)(n_base + p * 16 + r) * E_DIM + k0 + 8 * hf];
            v16bf b;
            #pragma unroll
            for (int j = 0; j < 8; ++j) { b[j] = brow[j]; b[8 + j] = brow[16 + j]; }
            acc[0][p] = __builtin_amdgcn_wmma_f32_16x16x32_bf16(
                false, a0, false, b, (short)0, acc[0][p], false, false);
            acc[1][p] = __builtin_amdgcn_wmma_f32_16x16x32_bf16(
                false, a1, false, b, (short)0, acc[1][p], false, false);
        }
    }
    __syncthreads();   // all reads of As complete before reuse

    // GELU(h2) tile into LDS (f32), per WMMA C/D layout
    float* h2s = (float*)smem;             // 32 x 512 f32 = 64 KB (exact reuse)
    #pragma unroll
    for (int p = 0; p < 4; ++p) {
        int n = n_base + p * 16 + r;
        float bias = b2[n];
        #pragma unroll
        for (int mt = 0; mt < 2; ++mt)
            #pragma unroll
            for (int v = 0; v < 8; ++v) {
                int m = 16 * mt + 8 * hf + v;
                h2s[m * H_DIM + n] = gelu_erf(acc[mt][p][v] + bias);
            }
    }
    __syncthreads();

    // Bias scores: 32 rows x 10 dots over 512 from LDS (320 pairs over 256 threads)
    const float th = thr[0];
    for (int t = tid; t < TM * NBIAS; t += 256) {
        int row = t / NBIAS;
        int i   = t - row * NBIAS;
        float s = b3[i];
        const float* hr = &h2s[row * H_DIM];
        const float* wr = &W3s[i * H_DIM];
        for (int j = 0; j < H_DIM; ++j) s += hr[j] * wr[j];
        float sig = 1.f / (1.f + expf(-s));
        if (sig > th) atomicOr(&mrow[row], 1 << i);
    }
    __syncthreads();

    // Compose per-row quaternion: Q = q_last (x) ... (x) q_first over set bits
    if (tid < TM) {
        float qw = 1.f, qx = 0.f, qy = 0.f, qz = 0.f;
        int bits = mrow[tid];
        #pragma unroll
        for (int i = 0; i < NBIAS; ++i) {
            if (bits & (1 << i)) {
                float w1 = quats[i * 4 + 0], x1 = quats[i * 4 + 1];
                float y1 = quats[i * 4 + 2], z1 = quats[i * 4 + 3];
                float w2 = qw, x2 = qx, y2 = qy, z2 = qz;
                qw = w1 * w2 - x1 * x2 - y1 * y2 - z1 * z2;
                qx = w1 * x2 + x1 * w2 + y1 * z2 - z1 * y2;
                qy = w1 * y2 - x1 * z2 + y1 * w2 + z1 * x2;
                qz = w1 * z2 + x1 * y2 - y1 * x2 + z1 * w2;
            }
        }
        QS[tid][0] = qw; QS[tid][1] = qx; QS[tid][2] = qy; QS[tid][3] = qz;
    }
    __syncthreads();

    // Apply: out = Q (x) x_quat, 32 rows x 256 quats, coalesced float4 stream
    const float4* xs = (const float4*)(x + m0 * E_DIM);
    float4* os = (float4*)(out + m0 * E_DIM);
    #pragma unroll
    for (int it = 0; it < 32; ++it) {
        int idx = tid + it * 256;       // 8192 float4
        int row = idx >> 8;             // 256 quats per row
        float4 p = xs[idx];
        float w1 = QS[row][0], x1 = QS[row][1], y1 = QS[row][2], z1 = QS[row][3];
        float4 o;
        o.x = w1 * p.x - x1 * p.y - y1 * p.z - z1 * p.w;
        o.y = w1 * p.y + x1 * p.x + y1 * p.w - z1 * p.z;
        o.z = w1 * p.z - x1 * p.w + y1 * p.x + z1 * p.y;
        o.w = w1 * p.w + x1 * p.z - y1 * p.y + z1 * p.x;
        os[idx] = o;
    }
}

extern "C" void kernel_launch(void* const* d_in, const int* in_sizes, int n_in,
                              void* d_out, int out_size, void* d_ws, size_t ws_size,
                              hipStream_t stream)
{
    (void)n_in; (void)out_size; (void)ws_size;
    const float* x     = (const float*)d_in[0];
    const float* W1    = (const float*)d_in[1];
    const float* b1    = (const float*)d_in[2];
    const float* ln_g  = (const float*)d_in[3];
    const float* ln_b  = (const float*)d_in[4];
    const float* W2    = (const float*)d_in[5];
    const float* b2    = (const float*)d_in[6];
    const float* W3    = (const float*)d_in[7];
    const float* b3    = (const float*)d_in[8];
    const float* quats = (const float*)d_in[9];
    const float* thr   = (const float*)d_in[10];
    float* out = (float*)d_out;

    const int BT = in_sizes[0] / E_DIM;   // B*T = 32768 rows

    // Workspace layout: W1 bf16 (2 MB) | W2 bf16 (1 MB) | h bf16 (BT*E*2 = 64 MB)
    char* ws = (char*)d_ws;
    bf16_t* W1b = (bf16_t*)ws;
    bf16_t* W2b = (bf16_t*)(ws + (size_t)2 * 1024 * 1024);
    bf16_t* hb  = (bf16_t*)(ws + (size_t)3 * 1024 * 1024);

    const size_t lds_tile = (size_t)TM * E_DIM * sizeof(bf16_t);   // 64 KB dynamic LDS

    cvt_weights<<<(E_DIM * E_DIM + 255) / 256, 256, 0, stream>>>(W1, W2, W1b, W2b);
    gemm1_ln_gelu<<<BT / TM, 256, lds_tile, stream>>>(x, W1b, b1, ln_g, ln_b, hb);
    gemm2_scores_quat<<<BT / TM, 256, lds_tile, stream>>>(hb, W2b, b2, W3, b3, quats, thr, x, out);
}
